// LigandModel_6030134084157
// MI455X (gfx1250) — matrix operand order repaired and checked by reading, not compiled, not run on hardware
//
#include <hip/hip_runtime.h>
#include <hip/hip_bf16.h>
#include <math.h>

// ---------------------------------------------------------------------------
// MI455X (gfx1250) implementation.
//  - All dense linear layers: v_wmma_f32_16x16x32_bf16, one wave = 16x32
//    output tile (2 WMMAs sharing one gathered A fragment).
//  - Activations stored bf16 in workspace; A fragment = two global_load_b128.
//  - Weights pre-converted once per launch into a bf16 fragment-layout arena;
//    B fragment = one contiguous 32B load.
//  - Concat segments processed as separate hoisted sub-loops (gather index +
//    row base computed once per segment, not per K step).
//  - Segment sums / scatter-mean: global_atomic_add_f32.
// ---------------------------------------------------------------------------

typedef __bf16 bf16_t;
typedef __attribute__((ext_vector_type(16))) __bf16 v16bf;
typedef __attribute__((ext_vector_type(8)))  __bf16 v8bf;
typedef __attribute__((ext_vector_type(8)))  float  v8f;

#define NNODES 131072
#define NEDGES 524288
#define NBATCH 4096
#define HDIM   128
#define EPSBN  1e-5f
#define ALPHA  0.3678f

static __device__ __forceinline__ float siluf(float x) {
    return x * (1.0f / (1.0f + __expf(-x)));
}

static __device__ __forceinline__ v8f wmma_bf16(v16bf a, v16bf b, v8f c) {
    return __builtin_amdgcn_wmma_f32_16x16x32_bf16(false, a, false, b,
                                                   (short)0, c, false, false);
}

// ---------------------------------------------------------------------------
// Weight prep: W[K,Nout] (f32) -> bf16 fragments.
// Fragment layout: Wf[((t*Kst + ks)*32 + lane)*16 + j]; tile t = columns
// t*16..t*16+15, step ks = K ks*32..ks*32+31; lanes 0-15 hold K +0..15 of
// column t*16+lane, lanes 16-31 hold K +16..31 of column t*16+lane-16.
// nTpad = 2*ceil(Nout/32) so the 16x32 GEMM tile always has both halves.
// ---------------------------------------------------------------------------
__global__ void k_prepw(const float* __restrict__ W, bf16_t* __restrict__ Wf,
                        int K, int Nout, int Kst, int nTpad) {
    long g = (long)blockIdx.x * blockDim.x + threadIdx.x;
    long total = (long)nTpad * Kst * 512;
    if (g >= total) return;
    int  j    = (int)(g & 15);
    int  lane = (int)((g >> 4) & 31);
    long fi   = g >> 9;                    // t*Kst + ks
    int  ks   = (int)(fi % Kst);
    int  t    = (int)(fi / Kst);
    int  half = lane >> 4, l15 = lane & 15;
    int  ncol = t * 16 + l15;
    int  k    = ks * 32 + half * 16 + j;
    float v = (ncol < Nout && k < K) ? W[(size_t)k * Nout + ncol] : 0.0f;
    Wf[g] = (bf16_t)v;
}

// ---------------------------------------------------------------------------
// Per-segment K sub-loop: gather index + row base hoisted, inner loop is
// loads + 2 WMMAs only. sb: segment stored as bf16.
// ---------------------------------------------------------------------------
static __device__ __forceinline__ void seg_mm(
    const void* __restrict__ sp, const int* __restrict__ si, int segk, int sb,
    int mrow, int r0, int r1,
    const bf16_t* __restrict__ wbase, int Kst, int& ks,
    v8f& acc0, v8f& acc1)
{
    if (segk <= 0) return;
    const int r = si ? si[mrow] : mrow;    // fused gather, once per segment
    if (sb) {
        const bf16_t* arow = (const bf16_t*)sp + (size_t)r * segk;
        int off = 0;
#pragma unroll 2
        for (; off + 32 <= segk; off += 32, ++ks) {
            v8bf lo = *(const v8bf*)(arow + off + r0);
            v8bf hi = *(const v8bf*)(arow + off + r1);
            v16bf a = __builtin_shufflevector(lo, hi,
                        0,1,2,3,4,5,6,7,8,9,10,11,12,13,14,15);
            v16bf b0 = *(const v16bf*)(wbase + ((size_t)ks << 9));
            v16bf b1 = *(const v16bf*)(wbase + ((size_t)(Kst + ks) << 9));
            acc0 = wmma_bf16(a, b0, acc0);
            acc1 = wmma_bf16(a, b1, acc1);
        }
        if (off < segk) {                  // ragged tail (K not mult of 32)
            v16bf a;
#pragma unroll
            for (int j = 0; j < 8; ++j) {
                int ka = off + r0 + j, kc = off + r1 + j;
                a[j]     = ka < segk ? arow[ka] : (bf16_t)0.0f;
                a[j + 8] = kc < segk ? arow[kc] : (bf16_t)0.0f;
            }
            v16bf b0 = *(const v16bf*)(wbase + ((size_t)ks << 9));
            v16bf b1 = *(const v16bf*)(wbase + ((size_t)(Kst + ks) << 9));
            acc0 = wmma_bf16(a, b0, acc0);
            acc1 = wmma_bf16(a, b1, acc1);
            ++ks;
        }
    } else {
        const float* arow = (const float*)sp + (size_t)r * segk;
        int off = 0;
#pragma unroll 2
        for (; off + 32 <= segk; off += 32, ++ks) {
            const float4* p0 = (const float4*)(arow + off + r0);
            const float4* p1 = (const float4*)(arow + off + r1);
            float4 q0 = p0[0], q1 = p0[1], q2 = p1[0], q3 = p1[1];
            v16bf a;
            a[0]  = (bf16_t)q0.x; a[1]  = (bf16_t)q0.y; a[2]  = (bf16_t)q0.z; a[3]  = (bf16_t)q0.w;
            a[4]  = (bf16_t)q1.x; a[5]  = (bf16_t)q1.y; a[6]  = (bf16_t)q1.z; a[7]  = (bf16_t)q1.w;
            a[8]  = (bf16_t)q2.x; a[9]  = (bf16_t)q2.y; a[10] = (bf16_t)q2.z; a[11] = (bf16_t)q2.w;
            a[12] = (bf16_t)q3.x; a[13] = (bf16_t)q3.y; a[14] = (bf16_t)q3.z; a[15] = (bf16_t)q3.w;
            v16bf b0 = *(const v16bf*)(wbase + ((size_t)ks << 9));
            v16bf b1 = *(const v16bf*)(wbase + ((size_t)(Kst + ks) << 9));
            acc0 = wmma_bf16(a, b0, acc0);
            acc1 = wmma_bf16(a, b1, acc1);
        }
        if (off < segk) {
            v16bf a;
#pragma unroll
            for (int j = 0; j < 8; ++j) {
                int ka = off + r0 + j, kc = off + r1 + j;
                a[j]     = (bf16_t)(ka < segk ? arow[ka] : 0.0f);
                a[j + 8] = (bf16_t)(kc < segk ? arow[kc] : 0.0f);
            }
            v16bf b0 = *(const v16bf*)(wbase + ((size_t)ks << 9));
            v16bf b1 = *(const v16bf*)(wbase + ((size_t)(Kst + ks) << 9));
            acc0 = wmma_bf16(a, b0, acc0);
            acc1 = wmma_bf16(a, b1, acc1);
            ++ks;
        }
    }
}

// ---------------------------------------------------------------------------
// Gathered GEMM: out[M,Nout] = act( bn( A @ W + bias ) + resid )
// A = concat of up to 3 segments. One wave = 16x32 output tile.
// act: 0=none 1=silu 2=relu. outbf: store bf16 (else f32). resid is bf16.
// nt2sh: log2(ceil(Nout/32)) -- #16x32 column tiles is a power of two.
// ---------------------------------------------------------------------------
__global__ __launch_bounds__(128)
void gemm_kernel(const void* __restrict__ s0p, const int* __restrict__ s0i, int k0,
                 const void* __restrict__ s1p, const int* __restrict__ s1i, int k1,
                 const void* __restrict__ s2p, const int* __restrict__ s2i, int k2,
                 int segbf,
                 const bf16_t* __restrict__ Wf, int Kst,
                 const float* __restrict__ bias,
                 const float* __restrict__ g,  const float* __restrict__ bb,
                 const float* __restrict__ bm, const float* __restrict__ bv,
                 const bf16_t* __restrict__ resid,
                 void* __restrict__ outp, int outbf,
                 int M, int Nout, int act, int nt2sh)
{
    const int tiles = (M >> 4) << nt2sh;
    const int tile  = blockIdx.x * 4 + (threadIdx.x >> 5);
    if (tile >= tiles) return;                 // wave-uniform exit (EXEC all-1)

    const int tm   = tile >> nt2sh;
    const int tn   = tile & ((1 << nt2sh) - 1);
    const int lane = threadIdx.x & 31;
    const int half = lane >> 4;
    const int l15  = lane & 15;
    const int mrow = (tm << 4) + l15;
    const int r0   = half ? 16 : 0;            // A run bases within a 32-K step
    const int r1   = r0 + 8;

    const bf16_t* wbase = Wf + (((size_t)(tn * 2) * Kst) << 9) + (lane << 4);

    v8f acc0 = {}, acc1 = {};
    int ks = 0;
    seg_mm(s0p, s0i, k0, segbf & 1,        mrow, r0, r1, wbase, Kst, ks, acc0, acc1);
    seg_mm(s1p, s1i, k1, (segbf >> 1) & 1, mrow, r0, r1, wbase, Kst, ks, acc0, acc1);
    seg_mm(s2p, s2i, k2, (segbf >> 2) & 1, mrow, r0, r1, wbase, Kst, ks, acc0, acc1);

    // epilogue: bias -> BN(eval) -> residual -> activation -> store
    auto epi = [&](v8f acc, int ncol) {
        const bool okc = (ncol < Nout);
        float bsv = 0.f, scale = 1.f, mv = 0.f, bvv = 0.f;
        if (okc) {
            if (bias) bsv = bias[ncol];
            if (g) { scale = g[ncol] * rsqrtf(bv[ncol] + EPSBN); mv = bm[ncol]; bvv = bb[ncol]; }
        }
#pragma unroll
        for (int rI = 0; rI < 8; ++rI) {
            int row = (tm << 4) + rI + half * 8;   // C/D layout: M = r + 8*half
            float y = acc[rI] + bsv;
            if (g) y = scale * (y - mv) + bvv;
            if (okc) {
                size_t idx = (size_t)row * Nout + ncol;
                if (resid) y += (float)resid[idx];
                if (act == 1)      y = siluf(y);
                else if (act == 2) y = fmaxf(y, 0.0f);
                if (outbf) ((bf16_t*)outp)[idx] = (bf16_t)y;
                else       ((float*)outp)[idx]  = y;
            }
        }
    };
    const int ncol0 = (tn << 5) + l15;
    epi(acc0, ncol0);
    epi(acc1, ncol0 + 16);
}

// ---------------------------------------------------------------------------
// Scatter / elementwise helpers
// ---------------------------------------------------------------------------
__global__ void k_count(const int* __restrict__ col, float* __restrict__ cnt, int E) {
    int e = blockIdx.x * blockDim.x + threadIdx.x;
    if (e < E) atomicAdd(&cnt[col[e]], 1.0f);
}

__global__ void k_scatter_add(const bf16_t* __restrict__ src, const int* __restrict__ col,
                              float* __restrict__ dst, int E, int C) {
    int g = blockIdx.x * blockDim.x + threadIdx.x;
    if (g >= E * C) return;
    int e = g / C, c = g - e * C;
    atomicAdd(&dst[(size_t)col[e] * C + c], (float)src[(size_t)e * C + c]);
}

__global__ void k_divcnt(float* __restrict__ agg, const float* __restrict__ cnt,
                         int Nn, int C) {
    int g = blockIdx.x * blockDim.x + threadIdx.x;
    if (g >= Nn * C) return;
    agg[g] /= fmaxf(cnt[g / C], 1.0f);
}

__global__ void k_dinv(const float* __restrict__ cnt, float* __restrict__ dinv, int Nn) {
    int i = blockIdx.x * blockDim.x + threadIdx.x;
    if (i < Nn) dinv[i] = rsqrtf(cnt[i] + 1.0f);
}

__global__ void k_wedge(const int* __restrict__ row, const int* __restrict__ col,
                        const float* __restrict__ dinv, float* __restrict__ w, int E) {
    int e = blockIdx.x * blockDim.x + threadIdx.x;
    if (e < E) w[e] = dinv[row[e]] * dinv[col[e]];
}

__global__ void k_scatter_gcn(const bf16_t* __restrict__ hn, const int* __restrict__ row,
                              const int* __restrict__ col, const float* __restrict__ w,
                              float* __restrict__ agg, int E) {
    int g = blockIdx.x * blockDim.x + threadIdx.x;
    if (g >= E * 32) return;
    int e = g >> 5, c = g & 31;
    atomicAdd(&agg[(size_t)col[e] * 32 + c], (float)hn[(size_t)row[e] * 32 + c] * w[e]);
}

__global__ void k_combine(const float* __restrict__ agg, const bf16_t* __restrict__ hn,
                          const float* __restrict__ dinv, bf16_t* __restrict__ hn2, int Nn) {
    int g = blockIdx.x * blockDim.x + threadIdx.x;
    if (g >= Nn * 32) return;
    int i = g >> 5;
    float sw = dinv[i] * dinv[i];
    float h  = (float)hn[g];
    hn2[g] = (bf16_t)((1.0f - ALPHA) * (agg[g] + sw * h) + ALPHA * h);
}

__global__ void k_pool(const bf16_t* __restrict__ x, const int* __restrict__ batch,
                       float* __restrict__ xp, int Nn) {
    int g = blockIdx.x * blockDim.x + threadIdx.x;
    if (g >= Nn * HDIM) return;
    int i = g >> 7, c = g & 127;
    atomicAdd(&xp[(size_t)batch[i] * HDIM + c], (float)x[g]);
}

// ---------------------------------------------------------------------------
// Host side
// ---------------------------------------------------------------------------
struct LinP { const float* W; const float* b; };
struct BnP  { const float* g; const float* b; const float* m; const float* v; };

static inline int gblk(long n) { return (int)((n + 255) / 256); }

static inline void launch_gemm(hipStream_t st,
    const void* s0p, const int* s0i, int k0,
    const void* s1p, const int* s1i, int k1,
    const void* s2p, const int* s2i, int k2,
    int segbf, const bf16_t* Wf, const float* bias, const BnP* bn,
    const bf16_t* resid, void* out, int outbf, int M, int Nout, int act)
{
    int K   = k0 + k1 + k2;
    int Kst = (K + 31) >> 5;
    int nT2 = (Nout + 31) >> 5;
    int nt2sh = 0; while ((1 << nt2sh) < nT2) ++nt2sh;   // nT2 is 1/4/8 here
    int tiles = (M >> 4) << nt2sh;
    int blocks = (tiles + 3) >> 2;                       // 4 waves / block
    gemm_kernel<<<blocks, 128, 0, st>>>(s0p, s0i, k0, s1p, s1i, k1, s2p, s2i, k2,
        segbf, Wf, Kst, bias,
        bn ? bn->g : nullptr, bn ? bn->b : nullptr,
        bn ? bn->m : nullptr, bn ? bn->v : nullptr,
        resid, out, outbf, M, Nout, act, nt2sh);
}

extern "C" void kernel_launch(void* const* d_in, const int* in_sizes, int n_in,
                              void* d_out, int out_size, void* d_ws, size_t ws_size,
                              hipStream_t stream) {
    (void)in_sizes; (void)n_in; (void)out_size; (void)ws_size;

    const float* x_in  = (const float*)d_in[0];          // [N, 64]
    const float* ea_in = (const float*)d_in[1];          // [E, 6]
    const int*   eidx  = (const int*)  d_in[2];          // [2, E]
    const int*   erow  = eidx;
    const int*   ecol  = eidx + NEDGES;
    const int*   batch = (const int*)  d_in[3];          // [N]
    const float* fp    = (const float*)d_in[4];          // [B, 1024]

    int pi = 5;
    auto P      = [&]() { return (const float*)d_in[pi++]; };
    auto getLin = [&]() { LinP l; l.W = P(); l.b = P(); return l; };
    auto getBn  = [&]() { BnP  b; b.g = P(); b.b = P(); b.m = P(); b.v = P(); return b; };

    LinP node_emb = getLin();
    LinP edge_emb = getLin();
    struct ConvP { LinP em; BnP ebn; LinP nm1; BnP nbn1; LinP nm2; BnP nbn2; } convs[3];
    for (int i = 0; i < 3; ++i) {
        convs[i].em  = getLin(); convs[i].ebn  = getBn();
        convs[i].nm1 = getLin(); convs[i].nbn1 = getBn();
        convs[i].nm2 = getLin(); convs[i].nbn2 = getBn();
    }
    struct ResP { LinP dn, de; BnP b1n, b1e; const float* gcnW; LinP un, ue; BnP b2n, b2e; } res[4];
    for (int i = 0; i < 4; ++i) {
        res[i].dn  = getLin(); res[i].de  = getLin();
        res[i].b1n = getBn();  res[i].b1e = getBn();
        res[i].gcnW = P();
        res[i].un  = getLin(); res[i].ue  = getLin();
        res[i].b2n = getBn();  res[i].b2e = getBn();
    }
    LinP encoder = getLin();
    LinP energy[3]; for (int i = 0; i < 3; ++i) energy[i] = getLin();
    LinP disc[6];   for (int i = 0; i < 6; ++i) disc[i]   = getLin();

    // ---------------- workspace (byte allocator, 256B aligned) ----------------
    char* base = (char*)d_ws;
    size_t off = 0;
    auto alloc = [&](size_t bytes) -> void* {
        bytes = (bytes + 255) & ~(size_t)255;
        void* p = base + off; off += bytes; return p;
    };
    bf16_t* xa    = (bf16_t*)alloc((size_t)NNODES * HDIM * 2);
    bf16_t* xb    = (bf16_t*)alloc((size_t)NNODES * HDIM * 2);
    bf16_t* ea0   = (bf16_t*)alloc((size_t)NEDGES * HDIM * 2);
    bf16_t* ea1   = (bf16_t*)alloc((size_t)NEDGES * HDIM * 2);
    float*  agg   = (float*) alloc((size_t)NNODES * HDIM * 4);
    float*  cnt   = (float*) alloc((size_t)NNODES * 4);
    float*  dinv  = (float*) alloc((size_t)NNODES * 4);
    float*  we    = (float*) alloc((size_t)NEDGES * 4);
    bf16_t* hn    = (bf16_t*)alloc((size_t)NNODES * 32 * 2);
    bf16_t* he    = (bf16_t*)alloc((size_t)NEDGES * 32 * 2);
    float*  agg32 = (float*) alloc((size_t)NNODES * 32 * 4);
    bf16_t* hn2   = (bf16_t*)alloc((size_t)NNODES * 32 * 2);
    float*  xpool = (float*) alloc((size_t)NBATCH * HDIM * 4);
    bf16_t* fpe   = (bf16_t*)alloc((size_t)NBATCH * HDIM * 2);
    bf16_t* h1    = (bf16_t*)alloc((size_t)NBATCH * 256 * 2);
    bf16_t* h2    = (bf16_t*)alloc((size_t)NBATCH * 256 * 2);

    // weight fragment arena: convert every W once, fragment layout
    auto prep = [&](const float* W, int K, int Nout) -> const bf16_t* {
        int Kst = (K + 31) >> 5;
        int nTpad = 2 * ((Nout + 31) >> 5);
        long total = (long)nTpad * Kst * 512;
        bf16_t* Wf = (bf16_t*)alloc((size_t)total * 2);
        k_prepw<<<gblk(total), 256, 0, stream>>>(W, Wf, K, Nout, Kst, nTpad);
        return Wf;
    };
    const bf16_t* w_nodeemb = prep(node_emb.W, 64, HDIM);
    const bf16_t* w_edgeemb = prep(edge_emb.W, 6, HDIM);
    const bf16_t *w_em[3], *w_nm1[3], *w_nm2[3];
    for (int i = 0; i < 3; ++i) {
        w_em[i]  = prep(convs[i].em.W, 3 * HDIM, HDIM);
        w_nm1[i] = prep(convs[i].nm1.W, 2 * HDIM, HDIM);
        w_nm2[i] = prep(convs[i].nm2.W, 2 * HDIM, HDIM);
    }
    const bf16_t *w_dn[4], *w_de[4], *w_gcn[4], *w_un[4], *w_ue[4];
    for (int i = 0; i < 4; ++i) {
        w_dn[i]  = prep(res[i].dn.W, HDIM, 32);
        w_de[i]  = prep(res[i].de.W, HDIM, 32);
        w_gcn[i] = prep(res[i].gcnW, 32, 32);
        w_un[i]  = prep(res[i].un.W, 32, HDIM);
        w_ue[i]  = prep(res[i].ue.W, 32, HDIM);
    }
    const bf16_t* w_enc = prep(encoder.W, 1024, HDIM);
    const bf16_t* w_en[3]; for (int i = 0; i < 3; ++i) w_en[i] = prep(energy[i].W, 256, i == 2 ? 4 : 256);
    const bf16_t* w_ds[6]; for (int i = 0; i < 6; ++i) w_ds[i] = prep(disc[i].W, 256, i == 5 ? 4 : 256);

    // ---------------- embeddings (f32 inputs -> bf16 activations) ----------------
    launch_gemm(stream, x_in, nullptr, 64, nullptr, nullptr, 0, nullptr, nullptr, 0,
                0, w_nodeemb, node_emb.b, nullptr, nullptr, xa, 1, NNODES, HDIM, 0);
    launch_gemm(stream, ea_in, nullptr, 6, nullptr, nullptr, 0, nullptr, nullptr, 0,
                0, w_edgeemb, edge_emb.b, nullptr, nullptr, ea0, 1, NEDGES, HDIM, 0);

    // in-degree counts (shared by scatter-mean and GCN normalization)
    hipMemsetAsync(cnt, 0, sizeof(float) * NNODES, stream);
    k_count<<<gblk(NEDGES), 256, 0, stream>>>(ecol, cnt, NEDGES);

    bf16_t* xc = xa; bf16_t* xo = xb;
    bf16_t* ec = ea0; bf16_t* eo = ea1;

    // ---------------- 3 MetaLayers ----------------
    for (int i = 0; i < 3; ++i) {
        ConvP& c = convs[i];
        // edge model: silu(bn(lin([x[row] | x[col] | ea])))
        launch_gemm(stream, xc, erow, HDIM, xc, ecol, HDIM, ec, nullptr, HDIM,
                    0x7, w_em[i], c.em.b, &c.ebn, nullptr, eo, 1, NEDGES, HDIM, 1);
        { bf16_t* t = ec; ec = eo; eo = t; }           // eo now free
        // node messages: silu(bn(lin([x[row] | ea]))) -> m in free buffer
        launch_gemm(stream, xc, erow, HDIM, ec, nullptr, HDIM, nullptr, nullptr, 0,
                    0x3, w_nm1[i], c.nm1.b, &c.nbn1, nullptr, eo, 1, NEDGES, HDIM, 1);
        // scatter-mean over col
        hipMemsetAsync(agg, 0, sizeof(float) * (size_t)NNODES * HDIM, stream);
        k_scatter_add<<<gblk((long)NEDGES * HDIM), 256, 0, stream>>>(eo, ecol, agg, NEDGES, HDIM);
        k_divcnt<<<gblk((long)NNODES * HDIM), 256, 0, stream>>>(agg, cnt, NNODES, HDIM);
        // node update: silu(bn(lin([x | agg])))  (agg is f32 -> segbf bit1=0)
        launch_gemm(stream, xc, nullptr, HDIM, agg, nullptr, HDIM, nullptr, nullptr, 0,
                    0x1, w_nm2[i], c.nm2.b, &c.nbn2, nullptr, xo, 1, NNODES, HDIM, 1);
        { bf16_t* t = xc; xc = xo; xo = t; }
    }

    // ---------------- GCN normalization ----------------
    k_dinv<<<gblk(NNODES), 256, 0, stream>>>(cnt, dinv, NNODES);
    k_wedge<<<gblk(NEDGES), 256, 0, stream>>>(erow, ecol, dinv, we, NEDGES);

    // ---------------- 4 residual GCN2 layers ----------------
    for (int i = 0; i < 4; ++i) {
        ResP& r = res[i];
        launch_gemm(stream, xc, nullptr, HDIM, nullptr, nullptr, 0, nullptr, nullptr, 0,
                    0x1, w_dn[i], r.dn.b, &r.b1n, nullptr, hn, 1, NNODES, 32, 1);
        launch_gemm(stream, ec, nullptr, HDIM, nullptr, nullptr, 0, nullptr, nullptr, 0,
                    0x1, w_de[i], r.de.b, &r.b1e, nullptr, he, 1, NEDGES, 32, 1);
        hipMemsetAsync(agg32, 0, sizeof(float) * (size_t)NNODES * 32, stream);
        k_scatter_gcn<<<gblk((long)NEDGES * 32), 256, 0, stream>>>(hn, erow, ecol, we, agg32, NEDGES);
        k_combine<<<gblk((long)NNODES * 32), 256, 0, stream>>>(agg32, hn, dinv, hn2, NNODES);
        // hn2 @ gcn_W  (no bias / bn / act) -> reuse hn buffer
        launch_gemm(stream, hn2, nullptr, 32, nullptr, nullptr, 0, nullptr, nullptr, 0,
                    0x1, w_gcn[i], nullptr, nullptr, nullptr, hn, 1, NNODES, 32, 0);
        // x = silu(bn(lin(un, .)) + x)
        launch_gemm(stream, hn, nullptr, 32, nullptr, nullptr, 0, nullptr, nullptr, 0,
                    0x1, w_un[i], r.un.b, &r.b2n, xc, xo, 1, NNODES, HDIM, 1);
        { bf16_t* t = xc; xc = xo; xo = t; }
        // ea = silu(bn(lin(ue, he)) + ea)
        launch_gemm(stream, he, nullptr, 32, nullptr, nullptr, 0, nullptr, nullptr, 0,
                    0x1, w_ue[i], r.ue.b, &r.b2e, ec, eo, 1, NEDGES, HDIM, 1);
        { bf16_t* t = ec; ec = eo; eo = t; }
    }

    // ---------------- pooling + fingerprint encoder ----------------
    hipMemsetAsync(xpool, 0, sizeof(float) * (size_t)NBATCH * HDIM, stream);
    k_pool<<<gblk((long)NNODES * HDIM), 256, 0, stream>>>(xc, batch, xpool, NNODES);
    launch_gemm(stream, fp, nullptr, 1024, nullptr, nullptr, 0, nullptr, nullptr, 0,
                0, w_enc, encoder.b, nullptr, nullptr, fpe, 1, NBATCH, HDIM, 1);

    float* out_f = (float*)d_out;

    // ---------------- energy head: 2x silu(2H->2H), relu(2H->4) ----------------
    launch_gemm(stream, xpool, nullptr, HDIM, fpe, nullptr, HDIM, nullptr, nullptr, 0,
                0x2, w_en[0], energy[0].b, nullptr, nullptr, h1, 1, NBATCH, 256, 1);
    launch_gemm(stream, h1, nullptr, 256, nullptr, nullptr, 0, nullptr, nullptr, 0,
                0x1, w_en[1], energy[1].b, nullptr, nullptr, h2, 1, NBATCH, 256, 1);
    launch_gemm(stream, h2, nullptr, 256, nullptr, nullptr, 0, nullptr, nullptr, 0,
                0x1, w_en[2], energy[2].b, nullptr, nullptr, out_f, 0, NBATCH, 4, 2);

    // ---------------- discriminator head: 5x silu(2H->2H), lin(2H->4) ----------
    launch_gemm(stream, xpool, nullptr, HDIM, fpe, nullptr, HDIM, nullptr, nullptr, 0,
                0x2, w_ds[0], disc[0].b, nullptr, nullptr, h1, 1, NBATCH, 256, 1);
    launch_gemm(stream, h1, nullptr, 256, nullptr, nullptr, 0, nullptr, nullptr, 0,
                0x1, w_ds[1], disc[1].b, nullptr, nullptr, h2, 1, NBATCH, 256, 1);
    launch_gemm(stream, h2, nullptr, 256, nullptr, nullptr, 0, nullptr, nullptr, 0,
                0x1, w_ds[2], disc[2].b, nullptr, nullptr, h1, 1, NBATCH, 256, 1);
    launch_gemm(stream, h1, nullptr, 256, nullptr, nullptr, 0, nullptr, nullptr, 0,
                0x1, w_ds[3], disc[3].b, nullptr, nullptr, h2, 1, NBATCH, 256, 1);
    launch_gemm(stream, h2, nullptr, 256, nullptr, nullptr, 0, nullptr, nullptr, 0,
                0x1, w_ds[4], disc[4].b, nullptr, nullptr, h1, 1, NBATCH, 256, 1);
    launch_gemm(stream, h1, nullptr, 256, nullptr, nullptr, 0, nullptr, nullptr, 0,
                0x1, w_ds[5], disc[5].b, nullptr, nullptr, out_f + (size_t)NBATCH * 4,
                0, NBATCH, 4, 0);
}
